// Block_11536282157829
// MI455X (gfx1250) — compile-verified, matrix operand order checked by reading, not run on hardware
//
#include <hip/hip_runtime.h>

// ---------------------------------------------------------------------------
// CDNA5 (gfx1250) wave32 WMMA implementation of the AIM adapter block.
// GEMMs: f16 WMMA 16x16x32, fp32 accumulate, software-pipelined global->LDS
// staging. Wide tiles (64x128, 8 WMMA/wave/k-step) for the big GEMMs; 64x64
// for the N=192 adapter down-projections.
// ---------------------------------------------------------------------------

typedef __attribute__((ext_vector_type(16))) _Float16 v16h;
typedef __attribute__((ext_vector_type(8)))  float    v8f;

__device__ __forceinline__ v8f wmma32(v16h a, v16h b, v8f c) {
  return __builtin_amdgcn_wmma_f32_16x16x32_f16(
      /*neg_a=*/false, a, /*neg_b=*/false, b,
      /*c_mod=*/(short)0, c, /*reuse_a=*/false, /*reuse_b=*/false);
}

#define NTOK 197
#define DIM  768
#define NSEQ 128
#define MT   12608           // 64 temporal seqs * 197
#define HEADS 12
#define QKVC 2304

// ---------------------------------------------------------------------------
// LayerNorm. One block per token row (768 cols, 256 threads, 3 elems/thread).
// mapped=1: input row is fetched from the interleaved [Bv,16,N,D] layout and
// written branch-ordered (temporal seqs first); also keeps a residual copy.
// ---------------------------------------------------------------------------
__global__ __launch_bounds__(256) void ln_kernel(
    const float* __restrict__ x, float* __restrict__ out,
    float* __restrict__ xcopy, const float* __restrict__ g,
    const float* __restrict__ b, int mapped) {
  int bt = blockIdx.x;
  int bs = bt / NTOK;
  int n  = bt - bs * NTOK;
  int gs = bs;
  if (mapped) {
    gs = (bs < 64) ? ((bs >> 3) * 16 + (bs & 7))
                   : (((bs - 64) >> 3) * 16 + 8 + ((bs - 64) & 7));
  }
  const float* xr = x + ((long)gs * NTOK + n) * DIM;
  float* orow = out + (long)bt * DIM;

  __shared__ float red[256];
  int t = threadIdx.x;
  float v0 = xr[t], v1 = xr[t + 256], v2 = xr[t + 512];
  red[t] = v0 + v1 + v2;
  __syncthreads();
  for (int off = 128; off > 0; off >>= 1) {
    if (t < off) red[t] += red[t + off];
    __syncthreads();
  }
  float mean = red[0] * (1.0f / 768.0f);
  __syncthreads();
  float d0 = v0 - mean, d1 = v1 - mean, d2 = v2 - mean;
  red[t] = d0 * d0 + d1 * d1 + d2 * d2;
  __syncthreads();
  for (int off = 128; off > 0; off >>= 1) {
    if (t < off) red[t] += red[t + off];
    __syncthreads();
  }
  float rstd = rsqrtf(red[0] * (1.0f / 768.0f) + 1e-5f);
  orow[t]       = d0 * rstd * g[t]       + b[t];
  orow[t + 256] = d1 * rstd * g[t + 256] + b[t + 256];
  orow[t + 512] = d2 * rstd * g[t + 512] + b[t + 512];
  if (xcopy) {
    float* c = xcopy + (long)bt * DIM;
    c[t] = v0; c[t + 256] = v1; c[t + 512] = v2;
  }
}

// ---------------------------------------------------------------------------
// Generic WMMA GEMM: out = epi(A[M,K] @ W[N,K]^T + bias) (+res1)(+res2)
// act: 0 = none, 1 = exact GELU, 2 = quickGELU. M%64==0, N%BN==0, K%32==0.
// BN=128: 4 waves, each computes 64x32 (8 WMMA / k-step, 6 fragment loads).
// BN=64 : 4 waves, each computes 32x32 (adapter down-proj, N=192).
// Software pipelined: next k-step's global loads are issued before the WMMAs.
// ---------------------------------------------------------------------------
#define LDT 48   // f16 LDS row stride (96B, 32B-aligned fragments)

template <int BN>
__global__ __launch_bounds__(128) void gemm_kernel(
    const float* __restrict__ A, const float* __restrict__ W,
    const float* __restrict__ bias, const float* __restrict__ res1,
    const float* __restrict__ res2, float* __restrict__ out,
    int M, int Nc, int K, int act) {
  constexpr int MI = (BN == 128) ? 4 : 2;   // m-subtiles per wave
  __shared__ __align__(32) _Float16 As[64 * LDT];
  __shared__ __align__(32) _Float16 Bs[BN * LDT];

  int tid = threadIdx.x;
  int lane = tid & 31, wave = tid >> 5;
  int nl = lane & 15, half = lane >> 4;
  int mBase = blockIdx.y * 64, nBase = blockIdx.x * BN;
  int wm = (BN == 128) ? 0 : (wave >> 1) * 32;
  int wn = (BN == 128) ? wave * 32 : (wave & 1) * 32;

  v8f acc[MI][2] = {};

  // global->register staging mapping
  int arow = tid >> 1;
  int aseg = (tid & 1) * 16;
  const float4* apg = reinterpret_cast<const float4*>(A + (long)(mBase + arow) * K + aseg);
  _Float16* asx = &As[arow * LDT + aseg];

  float4 a4[4];
  float4 b4[BN / 16];

  const float4* bpg;
  _Float16* bsx;
  if (BN == 128) {
    // each thread stages one full B row (32 k values)
    bpg = reinterpret_cast<const float4*>(W + (long)(nBase + tid) * K);
    bsx = &Bs[tid * LDT];
  } else {
    bpg = reinterpret_cast<const float4*>(W + (long)(nBase + arow) * K + aseg);
    bsx = &Bs[arow * LDT + aseg];
  }

  // prologue: load k0 = 0
#pragma unroll
  for (int i = 0; i < 4; i++) a4[i] = apg[i];
#pragma unroll
  for (int i = 0; i < BN / 16; i++) b4[i] = bpg[i];

  for (int k0 = 0; k0 < K; k0 += 32) {
    __syncthreads();  // previous iteration's fragment reads complete
#pragma unroll
    for (int i = 0; i < 4; i++) {
      asx[4 * i + 0] = (_Float16)a4[i].x; asx[4 * i + 1] = (_Float16)a4[i].y;
      asx[4 * i + 2] = (_Float16)a4[i].z; asx[4 * i + 3] = (_Float16)a4[i].w;
    }
#pragma unroll
    for (int i = 0; i < BN / 16; i++) {
      bsx[4 * i + 0] = (_Float16)b4[i].x; bsx[4 * i + 1] = (_Float16)b4[i].y;
      bsx[4 * i + 2] = (_Float16)b4[i].z; bsx[4 * i + 3] = (_Float16)b4[i].w;
    }
    __syncthreads();

    // issue next k-step's global loads; they fly behind the WMMAs below
    if (k0 + 32 < K) {
      int kq = (k0 + 32) >> 2;
#pragma unroll
      for (int i = 0; i < 4; i++) a4[i] = apg[kq + i];
#pragma unroll
      for (int i = 0; i < BN / 16; i++) b4[i] = bpg[kq + i];
    }

    v16h bf[2];
#pragma unroll
    for (int ni = 0; ni < 2; ni++)
      bf[ni] = *reinterpret_cast<const v16h*>(
          &Bs[(wn + ni * 16 + nl) * LDT + half * 16]);
#pragma unroll
    for (int mi = 0; mi < MI; mi++) {
      v16h af = *reinterpret_cast<const v16h*>(
          &As[(wm + mi * 16 + nl) * LDT + half * 16]);
#pragma unroll
      for (int ni = 0; ni < 2; ni++)
        acc[mi][ni] = wmma32(af, bf[ni], acc[mi][ni]);
    }
  }

#pragma unroll
  for (int mi = 0; mi < MI; mi++) {
#pragma unroll
    for (int ni = 0; ni < 2; ni++) {
      int col = nBase + wn + ni * 16 + nl;
      float bb = bias ? bias[col] : 0.0f;
#pragma unroll
      for (int r = 0; r < 8; r++) {
        int row = mBase + wm + mi * 16 + r + 8 * half;
        float v = acc[mi][ni][r] + bb;
        if (act == 1) v = 0.5f * v * (1.0f + erff(v * 0.70710678f));
        else if (act == 2) v = v / (1.0f + __expf(-1.702f * v));
        long oi = (long)row * Nc + col;
        if (res1) v += res1[oi];
        if (res2) v += res2[oi];
        out[oi] = v;
      }
    }
  }
}

// ---------------------------------------------------------------------------
// Attention: one block per (seq, head, qtile-group). 4 waves, each wave owns
// one 16-row Q tile. Scores (16x224) kept in 14 fully-unrolled WMMA
// accumulators (registers, no scratch); softmax via cross-lane shfl_xor;
// probs round-trip through per-wave LDS slab; AV via WMMA against V staged
// transposed (f16) in LDS.
// ---------------------------------------------------------------------------
#define NP 224   // keys padded to multiple of 32

__global__ __launch_bounds__(128) void attn_kernel(
    const float* __restrict__ qkv, float* __restrict__ o) {
  __shared__ __align__(32) _Float16 Vt[64 * NP];        // V transposed [d][key]
  __shared__ __align__(32) _Float16 probs[4 * 16 * NP]; // per-wave prob slab

  int tid = threadIdx.x;
  int lane = tid & 31, wave = tid >> 5;
  int nl = lane & 15, half = lane >> 4;
  int seq  = blockIdx.x / HEADS;
  int head = blockIdx.x - seq * HEADS;

  // Stage V transposed into LDS as f16 (zero pad rows >= 197)
  for (int r = tid; r < NP; r += 128) {
    if (r < NTOK) {
      const float4* vp = reinterpret_cast<const float4*>(
          qkv + ((long)(seq * NTOK + r)) * QKVC + 2 * DIM + head * 64);
#pragma unroll
      for (int c4 = 0; c4 < 16; c4++) {
        float4 v = vp[c4];
        Vt[(c4 * 4 + 0) * NP + r] = (_Float16)v.x;
        Vt[(c4 * 4 + 1) * NP + r] = (_Float16)v.y;
        Vt[(c4 * 4 + 2) * NP + r] = (_Float16)v.z;
        Vt[(c4 * 4 + 3) * NP + r] = (_Float16)v.w;
      }
    } else {
      for (int c = 0; c < 64; c++) Vt[c * NP + r] = (_Float16)0.0f;
    }
  }
  __syncthreads();

  int t = blockIdx.y * 4 + wave;
  if (t >= 13) return;

  // Q fragments straight from global (contiguous along head_dim)
  int qrow = t * 16 + nl; if (qrow > NTOK - 1) qrow = NTOK - 1;
  v16h aq[2];
#pragma unroll
  for (int kk = 0; kk < 2; kk++) {
    const float4* qp = reinterpret_cast<const float4*>(
        qkv + ((long)(seq * NTOK + qrow)) * QKVC + head * 64 + kk * 32 + half * 16);
#pragma unroll
    for (int i = 0; i < 4; i++) {
      float4 q4 = qp[i];
      aq[kk][4 * i + 0] = (_Float16)q4.x; aq[kk][4 * i + 1] = (_Float16)q4.y;
      aq[kk][4 * i + 2] = (_Float16)q4.z; aq[kk][4 * i + 3] = (_Float16)q4.w;
    }
  }

  // Scores = Q K^T : 14 key tiles x 2 k-steps. FULLY unrolled so sc[] stays
  // register-resident (partial unroll would demote it to scratch).
  v8f sc[14] = {};
#pragma unroll
  for (int j = 0; j < 14; j++) {
    int kr = j * 16 + nl; if (kr > NTOK - 1) kr = NTOK - 1;
#pragma unroll
    for (int kk = 0; kk < 2; kk++) {
      v16h bk;
      const float4* kp = reinterpret_cast<const float4*>(
          qkv + ((long)(seq * NTOK + kr)) * QKVC + DIM + head * 64 + kk * 32 + half * 16);
#pragma unroll
      for (int i = 0; i < 4; i++) {
        float4 k4 = kp[i];
        bk[4 * i + 0] = (_Float16)k4.x; bk[4 * i + 1] = (_Float16)k4.y;
        bk[4 * i + 2] = (_Float16)k4.z; bk[4 * i + 3] = (_Float16)k4.w;
      }
      sc[j] = wmma32(aq[kk], bk, sc[j]);
    }
  }

  // Softmax per row (rows r+8*half, columns spread over lane&15 and j)
  _Float16* pslab = &probs[wave * 16 * NP];
#pragma unroll
  for (int r = 0; r < 8; r++) {
    float m = -1e30f;
#pragma unroll
    for (int j = 0; j < 14; j++) {
      float v = sc[j][r] * 0.125f;          // hd^-0.5
      if (j * 16 + nl >= NTOK) v = -1e30f;  // mask padding
      sc[j][r] = v;
      m = fmaxf(m, v);
    }
    m = fmaxf(m, __shfl_xor(m, 1, 32));
    m = fmaxf(m, __shfl_xor(m, 2, 32));
    m = fmaxf(m, __shfl_xor(m, 4, 32));
    m = fmaxf(m, __shfl_xor(m, 8, 32));
    float s = 0.0f;
#pragma unroll
    for (int j = 0; j < 14; j++) {
      float e = __expf(sc[j][r] - m);
      sc[j][r] = e;
      s += e;
    }
    s += __shfl_xor(s, 1, 32);
    s += __shfl_xor(s, 2, 32);
    s += __shfl_xor(s, 4, 32);
    s += __shfl_xor(s, 8, 32);
    float inv = 1.0f / s;
    int prow = r + 8 * half;
#pragma unroll
    for (int j = 0; j < 14; j++)
      pslab[prow * NP + j * 16 + nl] = (_Float16)(sc[j][r] * inv);
  }
  // LDS ops from the same wave are kept in order (DScnt) -> safe to read back.

  // O = P V : 7 k-steps over 224 keys, 4 output column tiles
  v8f ao[4] = {};
#pragma unroll
  for (int kk = 0; kk < 7; kk++) {
    v16h ap = *reinterpret_cast<const v16h*>(&pslab[nl * NP + kk * 32 + half * 16]);
#pragma unroll
    for (int nt = 0; nt < 4; nt++) {
      v16h bv = *reinterpret_cast<const v16h*>(
          &Vt[(nt * 16 + nl) * NP + kk * 32 + half * 16]);
      ao[nt] = wmma32(ap, bv, ao[nt]);
    }
  }

#pragma unroll
  for (int nt = 0; nt < 4; nt++) {
#pragma unroll
    for (int r = 0; r < 8; r++) {
      int row = t * 16 + r + 8 * half;
      if (row < NTOK)
        o[((long)(seq * NTOK + row)) * DIM + head * 64 + nt * 16 + nl] = ao[nt][r];
    }
  }
}

// ---------------------------------------------------------------------------
__global__ void add3_kernel(const float* __restrict__ a, const float* __restrict__ b,
                            const float* __restrict__ c, float* __restrict__ o, long n) {
  long i = (long)blockIdx.x * 256 + threadIdx.x;
  if (i < n) o[i] = a[i] + b[i] + c[i];
}

__global__ void unmap_kernel(const float* __restrict__ in, float* __restrict__ out) {
  long i = (long)blockIdx.x * 256 + threadIdx.x;
  if (i >= (long)NSEQ * NTOK * DIM) return;
  long row = i / DIM;
  long col = i - row * DIM;
  int bt = (int)row;
  int bs = bt / NTOK;
  int n  = bt - bs * NTOK;
  int gs = (bs < 64) ? ((bs >> 3) * 16 + (bs & 7))
                     : (((bs - 64) >> 3) * 16 + 8 + ((bs - 64) & 7));
  out[((long)gs * NTOK + n) * DIM + col] = in[i];
}

// ---------------------------------------------------------------------------
extern "C" void kernel_launch(void* const* d_in, const int* in_sizes, int n_in,
                              void* d_out, int out_size, void* d_ws, size_t ws_size,
                              hipStream_t stream) {
  (void)in_sizes; (void)n_in; (void)out_size; (void)ws_size;
  const float* x      = (const float*)d_in[0];
  const float* qkv_w  = (const float*)d_in[1];
  const float* proj_w = (const float*)d_in[2];
  const float* proj_b = (const float*)d_in[3];
  const float* n1_g   = (const float*)d_in[4];
  const float* n1_b   = (const float*)d_in[5];
  const float* n2_g   = (const float*)d_in[6];
  const float* n2_b   = (const float*)d_in[7];
  const float* fc1_w  = (const float*)d_in[8];
  const float* fc1_b  = (const float*)d_in[9];
  const float* fc2_w  = (const float*)d_in[10];
  const float* fc2_b  = (const float*)d_in[11];
  const float* tab_dw = (const float*)d_in[12]; const float* tab_db = (const float*)d_in[13];
  const float* tab_uw = (const float*)d_in[14]; const float* tab_ub = (const float*)d_in[15];
  const float* sa_dw  = (const float*)d_in[16]; const float* sa_db  = (const float*)d_in[17];
  const float* sa_uw  = (const float*)d_in[18]; const float* sa_ub  = (const float*)d_in[19];
  const float* ta_dw  = (const float*)d_in[20]; const float* ta_db  = (const float*)d_in[21];
  const float* ta_uw  = (const float*)d_in[22]; const float* ta_ub  = (const float*)d_in[23];
  const float* sm_dw  = (const float*)d_in[24]; const float* sm_db  = (const float*)d_in[25];
  const float* sm_uw  = (const float*)d_in[26]; const float* sm_ub  = (const float*)d_in[27];
  const float* tm_dw  = (const float*)d_in[28]; const float* tm_db  = (const float*)d_in[29];
  const float* tm_uw  = (const float*)d_in[30]; const float* tm_ub  = (const float*)d_in[31];

  const long SZ    = (long)NSEQ * NTOK * DIM;   // 19,365,888
  const long SZT   = (long)MT * DIM;            //  9,682,944
  const long QS    = (long)NSEQ * NTOK * QKVC;  // 58,097,664
  const long BIGSZ = (long)NSEQ * NTOK * 3072;  // 77,463,552

  float* ws   = (float*)d_ws;
  float* XRES = ws;                 // branch-ordered residual copy of x
  float* BUFA = XRES + SZ;          // XN -> PRJ -> MLPO
  float* BUFB = BUFA + SZ;          // attn input -> X1
  float* BIG  = BUFB + SZ;          // QKV+AO -> MLP hidden -> final staging
  float* SAO  = BIG + BIGSZ;        // spatial attn adapter out
  float* XN2  = SAO + SZT;          // LN2 output
  float* BOT  = XN2 + SZ;           // adapter bottleneck scratch [12608,192]
  float* QKV  = BIG;
  float* AO   = BIG + QS;

  auto G64  = [](int M, int N) { return dim3((unsigned)(N / 64),  (unsigned)(M / 64)); };
  auto G128 = [](int M, int N) { return dim3((unsigned)(N / 128), (unsigned)(M / 64)); };

  // 1. LN1 (interleaved -> branch-ordered) + residual copy
  ln_kernel<<<NSEQ * NTOK, 256, 0, stream>>>(x, BUFA, XRES, n1_g, n1_b, 1);
  // 2-3. temporal pre-attn adapter (tab): down+GELU, up+bias -> attn input (temporal)
  gemm_kernel<64><<<G64(MT, 192), 128, 0, stream>>>(BUFA, tab_dw, tab_db, nullptr, nullptr, BOT, MT, 192, DIM, 1);
  gemm_kernel<128><<<G128(MT, DIM), 128, 0, stream>>>(BOT, tab_uw, tab_ub, nullptr, nullptr, BUFB, MT, DIM, 192, 0);
  // 4. attn input (spatial) = LN1 spatial
  hipMemcpyAsync(BUFB + SZT, BUFA + SZT, SZT * sizeof(float), hipMemcpyDeviceToDevice, stream);
  // 5-6. spatial attn adapter (sa) on LN1 spatial
  gemm_kernel<64><<<G64(MT, 192), 128, 0, stream>>>(BUFA + SZT, sa_dw, sa_db, nullptr, nullptr, BOT, MT, 192, DIM, 1);
  gemm_kernel<128><<<G128(MT, DIM), 128, 0, stream>>>(BOT, sa_uw, sa_ub, nullptr, nullptr, SAO, MT, DIM, 192, 0);
  // 7. QKV for both branches in one GEMM
  gemm_kernel<128><<<G128(NSEQ * NTOK, QKVC), 128, 0, stream>>>(BUFB, qkv_w, nullptr, nullptr, nullptr, QKV, NSEQ * NTOK, QKVC, DIM, 0);
  // 8. attention
  attn_kernel<<<dim3(NSEQ * HEADS, 4), 128, 0, stream>>>(QKV, AO);
  // 9. output projection
  gemm_kernel<128><<<G128(NSEQ * NTOK, DIM), 128, 0, stream>>>(AO, proj_w, proj_b, nullptr, nullptr, BUFA, NSEQ * NTOK, DIM, DIM, 0);
  // 10-11. temporal post-attn adapter (ta) + residual -> X1_t
  gemm_kernel<64><<<G64(MT, 192), 128, 0, stream>>>(BUFA, ta_dw, ta_db, nullptr, nullptr, BOT, MT, 192, DIM, 1);
  gemm_kernel<128><<<G128(MT, DIM), 128, 0, stream>>>(BOT, ta_uw, ta_ub, XRES, nullptr, BUFB, MT, DIM, 192, 0);
  // 12. X1_s = proj_s + sa_out + res_s
  add3_kernel<<<(unsigned)((SZT + 255) / 256), 256, 0, stream>>>(BUFA + SZT, SAO, XRES + SZT, BUFB + SZT, SZT);
  // 13. LN2
  ln_kernel<<<NSEQ * NTOK, 256, 0, stream>>>(BUFB, XN2, nullptr, n2_g, n2_b, 0);
  // 14-15. MLP (quickGELU)
  gemm_kernel<128><<<G128(NSEQ * NTOK, 3072), 128, 0, stream>>>(XN2, fc1_w, fc1_b, nullptr, nullptr, BIG, NSEQ * NTOK, 3072, DIM, 2);
  gemm_kernel<128><<<G128(NSEQ * NTOK, DIM), 128, 0, stream>>>(BIG, fc2_w, fc2_b, nullptr, nullptr, BUFA, NSEQ * NTOK, DIM, 3072, 0);
  // 16-17. temporal MLP adapter (tm) on mlp-out; final_t = tm_up + X1_t
  gemm_kernel<64><<<G64(MT, 192), 128, 0, stream>>>(BUFA, tm_dw, tm_db, nullptr, nullptr, BOT, MT, 192, DIM, 1);
  gemm_kernel<128><<<G128(MT, DIM), 128, 0, stream>>>(BOT, tm_uw, tm_ub, BUFB, nullptr, BIG, MT, DIM, 192, 0);
  // 18-19. spatial MLP adapter (sm) on LN2; final_s = sm_up + mlp_out_s + X1_s
  gemm_kernel<64><<<G64(MT, 192), 128, 0, stream>>>(XN2 + SZT, sm_dw, sm_db, nullptr, nullptr, BOT, MT, 192, DIM, 1);
  gemm_kernel<128><<<G128(MT, DIM), 128, 0, stream>>>(BOT, sm_uw, sm_ub, BUFA + SZT, BUFB + SZT, BIG + SZT, MT, DIM, 192, 0);
  // 20. branch-ordered -> interleaved output
  unmap_kernel<<<(unsigned)((SZ + 255) / 256), 256, 0, stream>>>(BIG, (float*)d_out);
}
